// Deduction_90546500534677
// MI455X (gfx1250) — compile-verified, matrix operand order checked by reading, not run on hardware
//
#include <hip/hip_runtime.h>
#include <hip/hip_bf16.h>

// ---------------- types / helpers ----------------
typedef __attribute__((ext_vector_type(16))) __bf16 v16bf;
typedef __attribute__((ext_vector_type(8)))  float  v8f;

#define GFLAG_ACC  1
#define GFLAG_GELU 2

__device__ __forceinline__ __bf16 f2bf(float f) {
    unsigned int u = __builtin_bit_cast(unsigned int, f);
    unsigned int r = u + 0x7FFFu + ((u >> 16) & 1u);   // round-to-nearest-even
    unsigned short h = (unsigned short)(r >> 16);
    return __builtin_bit_cast(__bf16, h);
}

__device__ __forceinline__ v8f v8zero() {
    v8f z;
#pragma unroll
    for (int e = 0; e < 8; ++e) z[e] = 0.0f;
    return z;
}

// Load one 16x32 (row x K) bf16 WMMA fragment for this lane.
// Lane layout (wave32): lanes 0-15 hold K=[0..7]+[16..23], lanes 16-31 K=[8..15]+[24..31].
__device__ __forceinline__ v16bf frag_ld(const __bf16* base, int half) {
    v16bf v;
#pragma unroll
    for (int e = 0; e < 8; ++e) {
        v[e]     = base[half * 8 + e];
        v[e + 8] = base[16 + half * 8 + e];
    }
    return v;
}

__device__ __forceinline__ float gelu_exact(float x) {
    return 0.5f * x * (1.0f + erff(x * 0.70710678118654752f));
}

// ---------------- generic strided WMMA GEMM ----------------
// C[m,n] = act( alpha * sum_k A[m*rsA + k*csA] * B[k*rsB + n*csB] + bias[n] ) (+C if ACC)
// Block tile 64x64, 128 threads = 4 waves (2x2), each wave 2x2 WMMA 16x16 tiles, K step 32.
// Interior tiles with a unit-stride dimension take vectorized float4 staging paths
// (global_load_b128 + global_prefetch_b8); edges / doubly-strided operands fall back
// to the guarded scalar path.
__global__ void __launch_bounds__(128)
k_gemm(const float* __restrict__ A, long rsA, long csA,
       const float* __restrict__ Bm, long rsB, long csB,
       const float* __restrict__ bias,
       float alpha, int flags,
       float* __restrict__ C, long rsC, long csC,
       int M, int Nn, int K)
{
    __shared__ __bf16 As[64][40];   // pitch 40 bf16 = 80B (16B multiple)
    __shared__ __bf16 Bs[64][40];   // Bs[n][k]

    const int tid  = threadIdx.x;
    const int wave = tid >> 5, lane = tid & 31;
    const int half = lane >> 4, lr = lane & 15;
    const int bm = blockIdx.y * 64, bn = blockIdx.x * 64;
    const int wm = (wave >> 1) * 32, wn = (wave & 1) * 32;

    const bool fullM = (bm + 64 <= M);
    const bool fullN = (bn + 64 <= Nn);

    v8f acc[2][2];
    acc[0][0] = v8zero(); acc[0][1] = v8zero();
    acc[1][0] = v8zero(); acc[1][1] = v8zero();

    for (int k0 = 0; k0 < K; k0 += 32) {
        const bool fullK = (k0 + 32 <= K);

        // ---- stage A tile (64 rows x 32 k) ----
        if (fullM && fullK && csA == 1) {
            // k-contiguous: one float4 per thread iteration, contiguous bf16x4 LDS store
#pragma unroll
            for (int i = tid; i < 64 * 8; i += 128) {
                int r = i >> 3, c4 = (i & 7) * 4;
                const float* p = A + (long)(bm + r) * rsA + (k0 + c4);
                if ((i & 7) == 0) __builtin_prefetch(p + 32, 0, 1);   // next K tile
                float4 v = *(const float4*)p;
                __bf16* dst = &As[r][c4];
                dst[0] = f2bf(v.x); dst[1] = f2bf(v.y); dst[2] = f2bf(v.z); dst[3] = f2bf(v.w);
            }
        } else {
            for (int i = tid; i < 64 * 32; i += 128) {
                int r = i >> 5, c = i & 31;
                int gm = bm + r, gk = k0 + c;
                float x = (gm < M && gk < K) ? A[(long)gm * rsA + (long)gk * csA] : 0.0f;
                As[r][c] = f2bf(x);
            }
        }

        // ---- stage B tile (64 n x 32 k), stored n-major ----
        if (fullN && fullK && rsB == 1) {
            // k-contiguous
#pragma unroll
            for (int i = tid; i < 64 * 8; i += 128) {
                int r = i >> 3, c4 = (i & 7) * 4;
                const float* p = Bm + (long)(bn + r) * csB + (k0 + c4);
                if ((i & 7) == 0) __builtin_prefetch(p + 32, 0, 1);
                float4 v = *(const float4*)p;
                __bf16* dst = &Bs[r][c4];
                dst[0] = f2bf(v.x); dst[1] = f2bf(v.y); dst[2] = f2bf(v.z); dst[3] = f2bf(v.w);
            }
        } else if (fullN && fullK && csB == 1) {
            // n-contiguous (weight matrices): float4 along n, 4 scalar LDS stores
#pragma unroll
            for (int i = tid; i < 32 * 16; i += 128) {
                int kk = i >> 4, n4 = (i & 15) * 4;
                const float* p = Bm + (long)(k0 + kk) * rsB + (bn + n4);
                if ((i & 15) == 0) __builtin_prefetch(p + 32 * rsB, 0, 1);
                float4 v = *(const float4*)p;
                Bs[n4 + 0][kk] = f2bf(v.x);
                Bs[n4 + 1][kk] = f2bf(v.y);
                Bs[n4 + 2][kk] = f2bf(v.z);
                Bs[n4 + 3][kk] = f2bf(v.w);
            }
        } else {
            for (int i = tid; i < 64 * 32; i += 128) {
                int r = i >> 5, c = i & 31;
                int gn = bn + r, gk = k0 + c;
                float x = (gn < Nn && gk < K) ? Bm[(long)gk * rsB + (long)gn * csB] : 0.0f;
                Bs[r][c] = f2bf(x);
            }
        }
        __syncthreads();

        v16bf a0 = frag_ld(&As[wm + lr][0], half);
        v16bf a1 = frag_ld(&As[wm + 16 + lr][0], half);
        v16bf b0 = frag_ld(&Bs[wn + lr][0], half);
        v16bf b1 = frag_ld(&Bs[wn + 16 + lr][0], half);

        acc[0][0] = __builtin_amdgcn_wmma_f32_16x16x32_bf16(false, a0, false, b0, (short)0, acc[0][0], false, false);
        acc[0][1] = __builtin_amdgcn_wmma_f32_16x16x32_bf16(false, a0, false, b1, (short)0, acc[0][1], false, false);
        acc[1][0] = __builtin_amdgcn_wmma_f32_16x16x32_bf16(false, a1, false, b0, (short)0, acc[1][0], false, false);
        acc[1][1] = __builtin_amdgcn_wmma_f32_16x16x32_bf16(false, a1, false, b1, (short)0, acc[1][1], false, false);
        __syncthreads();
    }

    // C/D layout: VGPR r -> lanes 0-15: (M=r, N=lane), lanes 16-31: (M=r+8, N=lane-16)
#pragma unroll
    for (int i = 0; i < 2; ++i)
#pragma unroll
        for (int j = 0; j < 2; ++j) {
            int gn = bn + wn + j * 16 + lr;
            if (gn >= Nn) continue;
            float bv = bias ? bias[gn] : 0.0f;
#pragma unroll
            for (int r = 0; r < 8; ++r) {
                int gm = bm + wm + i * 16 + half * 8 + r;
                if (gm >= M) continue;
                float val = acc[i][j][r] * alpha + bv;
                if (flags & GFLAG_GELU) val = gelu_exact(val);
                long off = (long)gm * rsC + (long)gn * csC;
                if (flags & GFLAG_ACC) val += C[off];
                C[off] = val;
            }
        }
}

// ---------------- generic 3-level strided softmax ----------------
// row rho -> (i0,i1,i2) with i0=rho/(d1*d2), i1=(rho/d2)%d1, i2=rho%d2
__global__ void k_softmax(const float* __restrict__ in, float* __restrict__ out, int n,
                          int d1, int d2,
                          long s0, long s1, long s2, long se,
                          long o0, long o1, long o2, long oe)
{
    int rho = blockIdx.x;
    int i0 = rho / (d1 * d2), rem = rho % (d1 * d2);
    int i1 = rem / d2, i2 = rem % d2;
    const float* ip = in + (long)i0 * s0 + (long)i1 * s1 + (long)i2 * s2;
    float* op = out + (long)i0 * o0 + (long)i1 * o1 + (long)i2 * o2;

    __shared__ float red[128];
    int tx = threadIdx.x, bs = blockDim.x;

    float m = -3.402823466e+38f;
    for (int j = tx; j < n; j += bs) m = fmaxf(m, ip[(long)j * se]);
    red[tx] = m; __syncthreads();
    for (int st = bs >> 1; st > 0; st >>= 1) { if (tx < st) red[tx] = fmaxf(red[tx], red[tx + st]); __syncthreads(); }
    m = red[0]; __syncthreads();

    float sum = 0.0f;
    for (int j = tx; j < n; j += bs) sum += __expf(ip[(long)j * se] - m);
    red[tx] = sum; __syncthreads();
    for (int st = bs >> 1; st > 0; st >>= 1) { if (tx < st) red[tx] += red[tx + st]; __syncthreads(); }
    float inv = 1.0f / red[0];

    for (int j = tx; j < n; j += bs) op[(long)j * oe] = __expf(ip[(long)j * se] - m) * inv;
}

// ---------------- residual-add LayerNorm (non-affine) ----------------
__global__ void k_ln_add2(const float* __restrict__ a, const float* __restrict__ b,
                          float* __restrict__ out, int width)
{
    long row = blockIdx.x;
    const float* ap = a + row * (long)width;
    const float* bp = b + row * (long)width;
    float* op = out + row * (long)width;

    __shared__ float red[256];
    int tx = threadIdx.x, bs = blockDim.x;

    float s = 0.0f;
    for (int j = tx; j < width; j += bs) s += ap[j] + bp[j];
    red[tx] = s; __syncthreads();
    for (int st = bs >> 1; st > 0; st >>= 1) { if (tx < st) red[tx] += red[tx + st]; __syncthreads(); }
    float mean = red[0] / (float)width; __syncthreads();

    float v = 0.0f;
    for (int j = tx; j < width; j += bs) { float d = ap[j] + bp[j] - mean; v += d * d; }
    red[tx] = v; __syncthreads();
    for (int st = bs >> 1; st > 0; st >>= 1) { if (tx < st) red[tx] += red[tx + st]; __syncthreads(); }
    float rstd = rsqrtf(red[0] / (float)width + 1e-12f);

    for (int j = tx; j < width; j += bs) op[j] = (ap[j] + bp[j] - mean) * rstd;
}

// ---------------- pt = swapaxes(b_mix, 1, 2) ----------------
__global__ void k_transpose_pt(const float* __restrict__ in, float* __restrict__ out)
{
    long t = (long)blockIdx.x * blockDim.x + threadIdx.x;
    long f = t * 4;                       // float4 granularity; 64-ch rows stay intact
    int c = (int)(f & 63);
    long q = f >> 6;
    int n = (int)(q & 511); q >>= 9;
    int m = (int)(q & 511); q >>= 9;      // q = b
    long ob = ((q * 512 + n) * 512 + m) * 64 + c;
    float4 v = *(const float4*)(in + f);
    *(float4*)(out + ob) = v;
}

// ---------------- fused binary MLP + pt@Ws + residual + LayerNorm ----------------
// Per 64-row block: h = gelu([bmix|pt] @ W1 + b1)  (kept in LDS, never hits HBM),
// acc = h @ W2 + pt @ Ws;  out = LN(bmix + acc + b2 + bs).
// 256 threads = 8 waves. Dynamic LDS: 182272 bytes.
#define MLP2_LDS_BYTES 182272
__global__ void __launch_bounds__(256)
k_mlp2(const float* __restrict__ bmix, const float* __restrict__ pt,
       const float* __restrict__ W1, const float* __restrict__ b1,
       const float* __restrict__ W2, const float* __restrict__ b2,
       const float* __restrict__ Ws, const float* __restrict__ bs,
       float* __restrict__ out)
{
    extern __shared__ char smem[];
    __bf16* sA1 = (__bf16*)smem;            // [64][72]   bmix tile
    __bf16* sA2 = sA1 + 64 * 72;            // [64][72]   pt tile
    __bf16* sW1 = sA2 + 64 * 72;            // [256][136] W1^T (n-major)
    __bf16* sW2 = sW1 + 256 * 136;          // [64][264]  W2^T
    __bf16* sWs = sW2 + 64 * 264;           // [64][72]   Ws^T
    __bf16* sH  = sWs + 64 * 72;            // [64][264]  hidden (m-major)
    float*  sO  = (float*)(sH + 64 * 264);  // [64][68]

    const int tid  = threadIdx.x;
    const int wave = tid >> 5, lane = tid & 31;
    const int half = lane >> 4, lr = lane & 15;
    const long row0 = (long)blockIdx.x * 64;

    for (int i = tid; i < 64 * 16; i += 256) {          // A tiles via float4
        int r = i >> 4, c4 = (i & 15) * 4;
        float4 va = *(const float4*)(bmix + (row0 + r) * 64 + c4);
        float4 vb = *(const float4*)(pt + (row0 + r) * 64 + c4);
        __bf16* d1 = &sA1[r * 72 + c4];
        __bf16* d2 = &sA2[r * 72 + c4];
        d1[0] = f2bf(va.x); d1[1] = f2bf(va.y); d1[2] = f2bf(va.z); d1[3] = f2bf(va.w);
        d2[0] = f2bf(vb.x); d2[1] = f2bf(vb.y); d2[2] = f2bf(vb.z); d2[3] = f2bf(vb.w);
    }
    for (int i = tid; i < 128 * 256; i += 256) { int k = i >> 8, n = i & 255; sW1[n * 136 + k] = f2bf(W1[k * 256 + n]); }
    for (int i = tid; i < 256 * 64;  i += 256) { int k = i >> 6, n = i & 63;  sW2[n * 264 + k] = f2bf(W2[k * 64 + n]); }
    for (int i = tid; i < 64 * 64;   i += 256) { int k = i >> 6, n = i & 63;  sWs[n * 72 + k]  = f2bf(Ws[k * 64 + n]); }
    __syncthreads();

    // ---- GEMM1: [64 x 128] @ [128 x 256]; waves 2x4, each 2x4 WMMA tiles
    {
        const int wm = (wave >> 2) * 32, wn = (wave & 3) * 64;
        v8f acc1[2][4];
#pragma unroll
        for (int i = 0; i < 2; ++i)
#pragma unroll
            for (int j = 0; j < 4; ++j) acc1[i][j] = v8zero();

        for (int k0 = 0; k0 < 128; k0 += 32) {
            const __bf16* Asrc = (k0 < 64) ? sA1 : sA2;   // concat([bmix, pt], -1)
            const int kk = k0 & 63;
            v16bf a[2], bfr[4];
#pragma unroll
            for (int i = 0; i < 2; ++i) a[i] = frag_ld(Asrc + (wm + i * 16 + lr) * 72 + kk, half);
#pragma unroll
            for (int j = 0; j < 4; ++j) bfr[j] = frag_ld(sW1 + (wn + j * 16 + lr) * 136 + k0, half);
#pragma unroll
            for (int i = 0; i < 2; ++i)
#pragma unroll
                for (int j = 0; j < 4; ++j)
                    acc1[i][j] = __builtin_amdgcn_wmma_f32_16x16x32_bf16(false, a[i], false, bfr[j], (short)0, acc1[i][j], false, false);
        }
        // bias + exact GELU -> bf16 hidden tile
#pragma unroll
        for (int i = 0; i < 2; ++i)
#pragma unroll
            for (int j = 0; j < 4; ++j) {
                int n = wn + j * 16 + lr;
                float bv = b1[n];
#pragma unroll
                for (int r = 0; r < 8; ++r) {
                    int m = wm + i * 16 + half * 8 + r;
                    sH[m * 264 + n] = f2bf(gelu_exact(acc1[i][j][r] + bv));
                }
            }
    }
    __syncthreads();

    // ---- GEMM2: [64 x 256] @ [256 x 64]  +  pt[64 x 64] @ Ws[64 x 64]
    {
        const int wm2 = (wave >> 1) * 16, wn2 = (wave & 1) * 32;
        v8f acc2[2];
        acc2[0] = v8zero(); acc2[1] = v8zero();

        for (int k0 = 0; k0 < 256; k0 += 32) {
            v16bf a = frag_ld(sH + (wm2 + lr) * 264 + k0, half);
#pragma unroll
            for (int j = 0; j < 2; ++j) {
                v16bf bb = frag_ld(sW2 + (wn2 + j * 16 + lr) * 264 + k0, half);
                acc2[j] = __builtin_amdgcn_wmma_f32_16x16x32_bf16(false, a, false, bb, (short)0, acc2[j], false, false);
            }
        }
        for (int k0 = 0; k0 < 64; k0 += 32) {
            v16bf a = frag_ld(sA2 + (wm2 + lr) * 72 + k0, half);
#pragma unroll
            for (int j = 0; j < 2; ++j) {
                v16bf bb = frag_ld(sWs + (wn2 + j * 16 + lr) * 72 + k0, half);
                acc2[j] = __builtin_amdgcn_wmma_f32_16x16x32_bf16(false, a, false, bb, (short)0, acc2[j], false, false);
            }
        }
        // + biases + residual b_mix
#pragma unroll
        for (int j = 0; j < 2; ++j) {
            int n = wn2 + j * 16 + lr;
            float bb = b2[n] + bs[n];
#pragma unroll
            for (int r = 0; r < 8; ++r) {
                int m = wm2 + half * 8 + r;
                sO[m * 68 + n] = acc2[j][r] + bb + bmix[(row0 + m) * 64 + n];
            }
        }
    }
    __syncthreads();

    // ---- LayerNorm over the 64-wide channel dim (one thread per row)
    if (tid < 64) {
        float s = 0.0f;
        for (int c = 0; c < 64; ++c) s += sO[tid * 68 + c];
        float mean = s * (1.0f / 64.0f);
        float v = 0.0f;
        for (int c = 0; c < 64; ++c) { float d = sO[tid * 68 + c] - mean; v += d * d; }
        float rstd = rsqrtf(v * (1.0f / 64.0f) + 1e-12f);
        for (int c = 0; c < 64; ++c) out[(row0 + tid) * 64 + c] = (sO[tid * 68 + c] - mean) * rstd;
    }
}

// ---------------- host orchestration ----------------
extern "C" void kernel_launch(void* const* d_in, const int* in_sizes, int n_in,
                              void* d_out, int out_size, void* d_ws, size_t ws_size,
                              hipStream_t stream)
{
    (void)in_sizes; (void)n_in; (void)out_size; (void)ws_size;
    const int Bb = 2, N = 512, D1 = 768, D2 = 64, H = 12, HD = 64, ALL = 768;
    const int I1 = 3072;
    const int BN = Bb * N;                 // 1024
    const long NN = (long)N * N;           // 262144
    const long R  = (long)Bb * NN;         // 524288

    const float* U       = (const float*)d_in[0];
    const float* B2      = (const float*)d_in[1];
    const float* join_Wv = (const float*)d_in[2];
    const float* join_bv = (const float*)d_in[3];
    const float* join_Ws = (const float*)d_in[4];
    const float* join_Ww = (const float*)d_in[5];
    const float* join_bw = (const float*)d_in[6];
    const float* cj_Wk   = (const float*)d_in[7];
    const float* cj_Wp   = (const float*)d_in[8];
    const float* cj_bp   = (const float*)d_in[9];
    const float* cj_Wo   = (const float*)d_in[10];
    const float* cj_bo   = (const float*)d_in[11];
    const float* as_Wqk  = (const float*)d_in[12];
    const float* as_bqk  = (const float*)d_in[13];
    const float* as_Ww   = (const float*)d_in[14];
    const float* as_bw   = (const float*)d_in[15];
    const float* tr_Wqk  = (const float*)d_in[16];
    const float* tr_Ww   = (const float*)d_in[17];
    const float* tr_bw   = (const float*)d_in[18];
    const float* b1_W1   = (const float*)d_in[19];
    const float* b1_b1   = (const float*)d_in[20];
    const float* b1_W2   = (const float*)d_in[21];
    const float* b1_b2   = (const float*)d_in[22];
    const float* b2_W1   = (const float*)d_in[23];
    const float* b2_b1   = (const float*)d_in[24];
    const float* b2_W2   = (const float*)d_in[25];
    const float* b2_b2   = (const float*)d_in[26];
    const float* b2_Ws   = (const float*)d_in[27];
    const float* b2_bs   = (const float*)d_in[28];

    float* out_u = (float*)d_out;                      // [2,512,768]
    float* out_b = out_u + (long)BN * D1;              // [2,512,512,64]

    // workspace carve-up (floats); big regions are reused across phases
    float* w = (float*)d_ws;
    float* buf_v    = w;                        // BN*768
    float* buf_kerl = buf_v    + (long)BN * ALL;
    float* buf_kers = buf_kerl + (long)BN * ALL;
    float* buf_qk   = buf_kers + (long)BN * ALL;           // BN*1536
    float* buf_uacc = buf_qk   + (long)BN * 2 * ALL;
    float* buf_umix = buf_uacc + (long)BN * D1;
    float* buf_h1   = buf_umix + (long)BN * D1;            // BN*3072
    float* buf_jp   = buf_h1   + (long)BN * I1;
    float* buf_cjp  = buf_jp   + (long)BN * ALL;
    float* buf_p12  = buf_cjp  + (long)BN * ALL;           // 2*12*NN (probs, then ao_pre)
    float* buf_big0 = buf_p12  + (long)Bb * H * NN;        // R*64 (prem -> q_raw -> to_pre)
    float* buf_big1 = buf_big0 + R * 64;                   // R*64 (tk)
    float* buf_big2 = buf_big1 + R * 64;                   // R*64 (tq -> pt)
    float* buf_bacc = buf_big2 + R * 64;                   // R*64 (ao + to)
    float* buf_bmix = buf_bacc + R * 64;                   // R*64

    auto gemm = [&](const float* A, long rsA, long csA,
                    const float* Bm, long rsB, long csB,
                    const float* bias, float alpha, int flags,
                    float* C, long rsC, long csC,
                    int M, int Nn, int K) {
        dim3 grid((Nn + 63) / 64, (M + 63) / 64);
        k_gemm<<<grid, dim3(128), 0, stream>>>(A, rsA, csA, Bm, rsB, csB, bias,
                                               alpha, flags, C, rsC, csC, M, Nn, K);
    };

    // ===== unary projections =====
    gemm(U, 768, 1, join_Wv, 768, 1, join_bv, 1.f, 0, buf_v, 768, 1, BN, ALL, D1);     // v
    gemm(U, 768, 1, cj_Wk, 768, 1, nullptr, 1.f, 0, buf_kerl, 768, 1, BN, ALL, D1);    // ker_lin
    gemm(U, 768, 1, as_Wqk, 1536, 1, as_bqk, 1.f, 0, buf_qk, 1536, 1, BN, 2 * ALL, D1);// qk

    // ===== JoinOperator =====
    for (int b = 0; b < Bb; ++b)   // sc[b,h,m,n] = B2 . join_Ws  (C scattered to [h][m*N+n])
        gemm(B2 + (long)b * NN * D2, 64, 1, join_Ws, 12, 1, nullptr, 1.f, 0,
             buf_p12 + (long)b * H * NN, 1, NN, (int)NN, H, D2);
    k_softmax<<<Bb * H * N, 128, 0, stream>>>(buf_p12, buf_p12, N, H, N,
        (long)H * NN, NN, N, 1, (long)H * NN, NN, N, 1);            // softmax over n, in-place
    for (int b = 0; b < Bb; ++b)
        for (int h = 0; h < H; ++h)    // jo_pre[b,m,h,d] = probs[b,h] @ v[b,:,h,:]
            gemm(buf_p12 + ((long)b * H + h) * NN, N, 1,
                 buf_v + (long)b * N * ALL + h * HD, 768, 1, nullptr, 1.f, 0,
                 buf_jp + (long)b * N * ALL + h * HD, 768, 1, N, HD, N);
    gemm(buf_jp, 768, 1, join_Ww, 768, 1, join_bw, 1.f, 0, buf_uacc, 768, 1, BN, D1, ALL);

    // ===== ConjugateJoinOperator =====
    k_softmax<<<Bb * HD * H, 128, 0, stream>>>(buf_kerl, buf_kers, N, HD, H,
        (long)N * ALL, 12, 1, 768, (long)N * ALL, (long)H * N, N, 1); // ker_sm[b,d,h,n]
    gemm(B2, 64, 1, cj_Wp, 64, 1, cj_bp, 1.f, 0, buf_big0, 64, 1, (int)R, D2, D2);     // prem[b,m,n,d]
    for (int b = 0; b < Bb; ++b)
        for (int d = 0; d < HD; ++d)   // cj_pre[b,n,d,h] = sum_m prem[b,d,m,n] * ker_sm[b,d,h,m]
            gemm(buf_big0 + (long)b * NN * D2 + d, 64, (long)N * 64,
                 buf_kers + ((long)b * HD + d) * H * N, 1, N, nullptr, 1.f, 0,
                 buf_cjp + (long)b * N * ALL + d * H, 768, 1, N, H, N);
    gemm(buf_cjp, 768, 1, cj_Wo, 768, 1, cj_bo, 1.f, GFLAG_ACC, buf_uacc, 768, 1, BN, D1, ALL);

    // ===== unary mixer + BooleanOperator MLP =====
    k_ln_add2<<<BN, 256, 0, stream>>>(U, buf_uacc, buf_umix, D1);
    gemm(buf_umix, 768, 1, b1_W1, 3072, 1, b1_b1, 1.f, GFLAG_GELU, buf_h1, 3072, 1, BN, I1, D1);
    gemm(buf_h1, 3072, 1, b1_W2, 768, 1, b1_b2, 1.f, 0, buf_uacc, 768, 1, BN, D1, I1);
    k_ln_add2<<<BN, 256, 0, stream>>>(buf_umix, buf_uacc, out_u, D1);

    // ===== AssociativeOperator =====
    for (int b = 0; b < Bb; ++b)
        for (int h = 0; h < H; ++h)    // ao_pre[b,m,n,h] = (q_h @ k_h^T)/8
            gemm(buf_qk + (long)b * N * 1536 + h * HD, 1536, 1,
                 buf_qk + (long)b * N * 1536 + ALL + h * HD, 1, 1536, nullptr, 0.125f, 0,
                 buf_p12 + (long)b * NN * H + h, (long)N * H, H, N, N, HD);
    gemm(buf_p12, 12, 1, as_Ww, 64, 1, as_bw, 1.f, 0, buf_bacc, 64, 1, (int)R, D2, H);

    // ===== CompositionOperator =====
    gemm(B2, 64, 1, tr_Wqk, 128, 1, nullptr, 1.f, 0, buf_big0, 64, 1, (int)R, D2, D2);      // q_raw
    gemm(B2, 64, 1, tr_Wqk + 64, 128, 1, nullptr, 1.f, 0, buf_big1, 64, 1, (int)R, D2, D2); // tk
    k_softmax<<<Bb * D2 * N, 128, 0, stream>>>(buf_big0, buf_big2, N, D2, N,
        NN * 64, 1, (long)N * 64, 64, 64 * NN, NN, N, 1);           // tq[b,c,m,n]
    for (int b = 0; b < Bb; ++b)
        for (int c = 0; c < D2; ++c)   // to_pre[b,m,k,c] = tq[b,c] @ tk[b,c]
            gemm(buf_big2 + ((long)b * D2 + c) * NN, N, 1,
                 buf_big1 + (long)b * NN * D2 + c, (long)N * 64, 64, nullptr, 1.f, 0,
                 buf_big0 + (long)b * NN * D2 + c, (long)N * 64, 64, N, N, N);
    gemm(buf_big0, 64, 1, tr_Ww, 64, 1, tr_bw, 1.f, GFLAG_ACC, buf_bacc, 64, 1, (int)R, D2, D2);

    // ===== binary mixer + fused MLP2 / pt@Ws / LayerNorm =====
    k_ln_add2<<<dim3((unsigned)R), 64, 0, stream>>>(B2, buf_bacc, buf_bmix, D2);
    k_transpose_pt<<<dim3((unsigned)(R * 64 / 4 / 256)), 256, 0, stream>>>(buf_bmix, buf_big2); // pt
    k_mlp2<<<dim3((unsigned)(R / 64)), 256, MLP2_LDS_BYTES, stream>>>(
        buf_bmix, buf_big2, b2_W1, b2_b1, b2_W2, b2_b2, b2_Ws, b2_bs, out_b);
}
